// GAT_GNN_35579509080109
// MI455X (gfx1250) — matrix operand order repaired and checked by reading, not compiled, not run on hardware
//
#include <hip/hip_runtime.h>

typedef __attribute__((ext_vector_type(16))) _Float16 v16h;
typedef __attribute__((ext_vector_type(8)))  _Float16 v8h;
typedef __attribute__((ext_vector_type(8)))  float    v8f;

#define DIMD 256
#define NEG_SLOPE 0.2f

// ---------------- activation f32 -> f16 ----------------
__global__ void k_f32_to_f16(const float* __restrict__ src, _Float16* __restrict__ dst, int n) {
  int i = blockIdx.x * blockDim.x + threadIdx.x;
  if (i < n) dst[i] = (_Float16)src[i];
}

__global__ void k_f32_to_f16_relu(const float* __restrict__ src, _Float16* __restrict__ dst, int n) {
  int i = blockIdx.x * blockDim.x + threadIdx.x;
  if (i < n) { float v = src[i]; dst[i] = (_Float16)(v > 0.f ? v : 0.f); }
}

// ---------------- pack weight B[256x256] (f32) into WMMA fragment layout (f16) ----------------
// Fragment order: [nt(16)][kt(8)][lane(32)][16 halves], so the GEMM loads each lane's
// 32-byte fragment contiguously (2x global_load_b128, coalesced across the wave).
// Contiguous k order, same as A frag: lane half hf covers K rows kb..kb+15 (pairs per VGPR), col = n0+(lane&15).
// fold=1: B = W3[0:256,:] + W3[256:512,:]  (relu(concat(h,h)) @ W3 == relu(h) @ folded)
__global__ void k_pack_b(const float* __restrict__ W, _Float16* __restrict__ out, int fold) {
  int tid = blockIdx.x * blockDim.x + threadIdx.x;   // 16*8*32 = 4096 threads
  if (tid >= 16 * 8 * 32) return;
  int lane = tid & 31;
  int kt   = (tid >> 5) & 7;
  int nt   = tid >> 8;
  int col  = nt * 16 + (lane & 15);
  int kb   = kt * 32 + (lane >> 4) * 16;
  _Float16* o = out + (size_t)tid * 16;
#pragma unroll
  for (int j = 0; j < 16; ++j) {
    float v = W[(kb + j) * DIMD + col];
    if (fold) v += W[(kb + j) * DIMD + col + DIMD * DIMD];
    o[j] = (_Float16)v;
  }
}

// ---------------- WMMA GEMM: C[M,256] = A[M,256](f16) * Bpacked(f16), f32 out ----------------
// Block = 128 threads (4 waves). Block computes 64 x 128 of C.
// The 64x256 A panel is staged in LDS once via async loads (ASYNCcnt) and shared by all waves;
// each wave computes a 64x32 slab (2 n-tiles x 4 m-tiles = 8 accumulators).
__global__ void __launch_bounds__(128) k_gemm_wmma(const _Float16* __restrict__ A,
                                                   const _Float16* __restrict__ Bp,
                                                   float* __restrict__ C, int M) {
  __shared__ __align__(16) _Float16 tileA[64 * DIMD];   // 32 KB
  const int tid = threadIdx.x;
  const int m0  = blockIdx.x * 64;

  // ---- async stage: 64 rows x 512B = 2048 x 16B chunks, 16 per thread, coalesced ----
  unsigned ldsbase = (unsigned)(uintptr_t)tileA;        // low 32 bits of LDS aperture = LDS offset
#pragma unroll
  for (int i = 0; i < 16; ++i) {
    int g    = i * 128 + tid;          // chunk id
    int lrow = g >> 5;                 // 32 chunks per row
    int c8   = (g & 31) * 8;           // column start in halves
    int grow = m0 + lrow; if (grow >= M) grow = M - 1;   // clamp tail rows
    const _Float16* gp = A + (size_t)grow * DIMD + c8;
    unsigned lo = ldsbase + (unsigned)g * 16u;
    asm volatile("global_load_async_to_lds_b128 %0, %1, off" :: "v"(lo), "v"(gp) : "memory");
  }
  asm volatile("s_wait_asynccnt 0x0" ::: "memory");
  __syncthreads();

  const int w    = tid >> 5;
  const int lane = tid & 31;
  const int r    = lane & 15;
  const int hf   = lane >> 4;
  const int ntBase = blockIdx.y * 8 + w * 2;   // this wave's two n-tiles

  v8f acc[2][4];
#pragma unroll
  for (int q = 0; q < 2; ++q)
#pragma unroll
    for (int t = 0; t < 4; ++t) acc[q][t] = v8f{};

  for (int kt = 0; kt < 8; ++kt) {
    v16h bf0 = *(const v16h*)(Bp + ((size_t)(ntBase * 8 + kt) * 32 + lane) * 16);
    v16h bf1 = *(const v16h*)(Bp + ((size_t)((ntBase + 1) * 8 + kt) * 32 + lane) * 16);
    const int ka = kt * 32 + hf * 16;   // A frag: e[0..15]=K ka..ka+15
#pragma unroll
    for (int t = 0; t < 4; ++t) {
      const _Float16* ap = tileA + ((t * 16 + r) * DIMD + ka);
      v8h a0 = *(const v8h*)(ap);
      v8h a1 = *(const v8h*)(ap + 8);
      v16h af;
#pragma unroll
      for (int j = 0; j < 8; ++j) { af[j] = a0[j]; af[j + 8] = a1[j]; }
      acc[0][t] = __builtin_amdgcn_wmma_f32_16x16x32_f16(
          false, af, false, bf0, (short)0, acc[0][t], false, false);
      acc[1][t] = __builtin_amdgcn_wmma_f32_16x16x32_f16(
          false, af, false, bf1, (short)0, acc[1][t], false, false);
    }
  }
  // C layout: VGPR rr -> row hf*8+rr, col = lane&15
#pragma unroll
  for (int q = 0; q < 2; ++q) {
    const int col = (ntBase + q) * 16 + r;
#pragma unroll
    for (int t = 0; t < 4; ++t) {
#pragma unroll
      for (int rr = 0; rr < 8; ++rr) {
        int grow = m0 + t * 16 + hf * 8 + rr;
        if (grow < M) C[(size_t)grow * DIMD + col] = acc[q][t][rr];
      }
    }
  }
}

// ---------------- attention scores: e_src = hw @ asrc, e_dst = hw @ adst ----------------
__global__ void k_scores(const float* __restrict__ hw, const float* __restrict__ asrc,
                         const float* __restrict__ adst, float* __restrict__ es,
                         float* __restrict__ ed, int n) {
  int lane = threadIdx.x & 31;
  int node = blockIdx.x * (blockDim.x >> 5) + (threadIdx.x >> 5);
  if (node >= n) return;
  const float* row = hw + (size_t)node * DIMD + lane * 8;
  float s = 0.f, d = 0.f;
#pragma unroll
  for (int j = 0; j < 8; ++j) {
    float v = row[j];
    s += v * asrc[lane * 8 + j];
    d += v * adst[lane * 8 + j];
  }
#pragma unroll
  for (int m = 16; m >= 1; m >>= 1) {
    s += __shfl_xor(s, m, 32);
    d += __shfl_xor(d, m, 32);
  }
  if (lane == 0) { es[node] = s; ed[node] = d; }
}

// ---------------- per-layer init: out = bias (broadcast), denom = 0, mkey = -inf key ----------------
__global__ void k_init_layer(float* __restrict__ out, const float* __restrict__ bias,
                             float* __restrict__ denom, unsigned* __restrict__ mkey, int n) {
  int i = blockIdx.x * blockDim.x + threadIdx.x;
  if (i >= n * DIMD) return;
  int d = i & (DIMD - 1);
  out[i] = bias[d];
  if (d == 0) { denom[i >> 8] = 0.f; mkey[i >> 8] = 0u; }
}

// order-preserving float<->uint key for atomicMax on signed floats
__device__ __forceinline__ unsigned f2key(float f) {
  unsigned u = __float_as_uint(f);
  return (u & 0x80000000u) ? ~u : (u | 0x80000000u);
}
__device__ __forceinline__ float key2f(unsigned k) {
  unsigned u = (k & 0x80000000u) ? (k & 0x7FFFFFFFu) : ~k;
  return __uint_as_float(u);
}

// ---------------- edge pass 1: e = leaky_relu(es[src]+ed[dst]); segment max ----------------
__global__ void k_edge_max(const long long* __restrict__ ei, const float* __restrict__ es,
                           const float* __restrict__ ed, float* __restrict__ ebuf,
                           unsigned* __restrict__ mkey, int ecount) {
  int e = blockIdx.x * blockDim.x + threadIdx.x;
  if (e >= ecount) return;
  int s = (int)ei[e];
  int t = (int)ei[ecount + e];
  float v = es[s] + ed[t];
  v = v > 0.f ? v : NEG_SLOPE * v;
  ebuf[e] = v;
  atomicMax(&mkey[t], f2key(v));
}

// ---------------- edge pass 2: ee = exp(e - m[dst]); segment sum ----------------
__global__ void k_edge_exp(const long long* __restrict__ ei, const float* __restrict__ ebuf,
                           const unsigned* __restrict__ mkey, float* __restrict__ eebuf,
                           float* __restrict__ denom, int ecount) {
  int e = blockIdx.x * blockDim.x + threadIdx.x;
  if (e >= ecount) return;
  int t = (int)ei[ecount + e];
  float ee = expf(ebuf[e] - key2f(mkey[t]));
  eebuf[e] = ee;
  atomicAdd(&denom[t], ee);
}

// ---------------- edge pass 3: out[dst] += (ee/denom[dst]) * hw[src] ----------------
// One wave per edge: coalesced 1KB row gather, 8 float atomics per lane (L2-resident).
__global__ void k_edge_aggr(const long long* __restrict__ ei, const float* __restrict__ eebuf,
                            const float* __restrict__ denom, const float* __restrict__ hw,
                            float* __restrict__ out, int ecount) {
  int lane = threadIdx.x & 31;
  int e = blockIdx.x * (blockDim.x >> 5) + (threadIdx.x >> 5);
  if (e >= ecount) return;
  int s = (int)ei[e];
  int t = (int)ei[ecount + e];
  float alpha = eebuf[e] / denom[t];
  const float* srow = hw + (size_t)s * DIMD + lane * 8;
  float* drow = out + (size_t)t * DIMD + lane * 8;
  __builtin_prefetch(srow, 0, 1);   // global_prefetch_b8
#pragma unroll
  for (int j = 0; j < 8; ++j) atomicAdd(&drow[j], alpha * srow[j]);
}

// ---------------- host launcher ----------------
extern "C" void kernel_launch(void* const* d_in, const int* in_sizes, int n_in,
                              void* d_out, int out_size, void* d_ws, size_t ws_size,
                              hipStream_t stream) {
  (void)n_in; (void)out_size; (void)ws_size;
  const float*     x     = (const float*)d_in[0];
  const long long* ei    = (const long long*)d_in[1];   // int64 edge_index [2,E]
  const float*     W1    = (const float*)d_in[2];
  const float*     W2    = (const float*)d_in[3];
  const float*     Ws    = (const float*)d_in[4];       // [6,D,D]
  const float*     a_src = (const float*)d_in[5];       // [6,D]
  const float*     a_dst = (const float*)d_in[6];       // [6,D]
  const float*     bias  = (const float*)d_in[7];       // [6,D]
  const float*     W3    = (const float*)d_in[8];       // [2D,OUT]
  float* out = (float*)d_out;

  const int N = in_sizes[0] / DIMD;    // 20000
  const int E = in_sizes[1] / 2;       // 320000

  // carve workspace (256B aligned regions)
  char* p = (char*)d_ws;
  auto carve = [&](size_t bytes) { char* q = p; p += (bytes + 255) & ~(size_t)255; return q; };
  float*    hA    = (float*)carve((size_t)N * DIMD * 4);       // h (layer in/out accumulator)
  float*    hwb   = (float*)carve((size_t)N * DIMD * 4);       // hw = h @ W
  _Float16* Af    = (_Float16*)carve((size_t)N * DIMD * 2);    // f16 staging of activations
  _Float16* Bf    = (_Float16*)carve((size_t)DIMD * DIMD * 2); // packed f16 weight fragments
  float*    es    = (float*)carve((size_t)N * 4);
  float*    ed    = (float*)carve((size_t)N * 4);
  float*    denom = (float*)carve((size_t)N * 4);
  unsigned* mkey  = (unsigned*)carve((size_t)N * 4);
  float*    ebuf  = (float*)carve((size_t)E * 4);
  float*    eebuf = (float*)carve((size_t)E * 4);

  dim3 b256(256), bGemm(128);
  dim3 gConv((N * DIMD + 255) / 256);
  dim3 gPack(16);                       // 4096 pack threads
  dim3 gGemm((N + 63) / 64, 2);         // 64x128 C per block
  dim3 gScore((N + 7) / 8);
  dim3 gEdge((E + 255) / 256);
  dim3 gAggr((E + 7) / 8);

  // h = (x @ W1) @ W2
  k_f32_to_f16<<<gConv, b256, 0, stream>>>(x, Af, N * DIMD);
  k_pack_b<<<gPack, b256, 0, stream>>>(W1, Bf, 0);
  k_gemm_wmma<<<gGemm, bGemm, 0, stream>>>(Af, Bf, hwb, N);
  k_f32_to_f16<<<gConv, b256, 0, stream>>>(hwb, Af, N * DIMD);
  k_pack_b<<<gPack, b256, 0, stream>>>(W2, Bf, 0);
  k_gemm_wmma<<<gGemm, bGemm, 0, stream>>>(Af, Bf, hA, N);

  // 6 GAT layers
  for (int l = 0; l < 6; ++l) {
    k_f32_to_f16<<<gConv, b256, 0, stream>>>(hA, Af, N * DIMD);
    k_pack_b<<<gPack, b256, 0, stream>>>(Ws + (size_t)l * DIMD * DIMD, Bf, 0);
    k_gemm_wmma<<<gGemm, bGemm, 0, stream>>>(Af, Bf, hwb, N);
    k_scores<<<gScore, b256, 0, stream>>>(hwb, a_src + l * DIMD, a_dst + l * DIMD, es, ed, N);
    k_init_layer<<<gConv, b256, 0, stream>>>(hA, bias + l * DIMD, denom, mkey, N);
    k_edge_max<<<gEdge, b256, 0, stream>>>(ei, es, ed, ebuf, mkey, E);
    k_edge_exp<<<gEdge, b256, 0, stream>>>(ei, ebuf, mkey, eebuf, denom, E);
    k_edge_aggr<<<gAggr, b256, 0, stream>>>(ei, eebuf, denom, hwb, hA, E);
  }

  // out = relu(h) @ (W3_top + W3_bot)
  k_pack_b<<<gPack, b256, 0, stream>>>(W3, Bf, 1);
  k_f32_to_f16_relu<<<gConv, b256, 0, stream>>>(hA, Af, N * DIMD);
  k_gemm_wmma<<<gGemm, bGemm, 0, stream>>>(Af, Bf, out, N);
}
